// YOLOv1Loss_26319559590103
// MI455X (gfx1250) — compile-verified, matrix operand order checked by reading, not run on hardware
//
#include <hip/hip_runtime.h>
#include <hip/hip_fp16.h>

// ---- problem constants (must match reference) ----
#define S_GRID   7
#define B_ANCH   2
#define C_CLS    80
#define NCH      (B_ANCH * 5 + C_CLS)      // 90 channels
#define N_GT     32
#define LC_W     5.0f
#define LN_W     0.5f
#define CELLS    (S_GRID * S_GRID)         // 49
#define PRED_ELEMS (CELLS * NCH)           // 4410 floats per batch element

typedef __attribute__((ext_vector_type(16))) _Float16 v16h;
typedef __attribute__((ext_vector_type(8)))  float    v8f;

// async-to-LDS b64 builtin wants pointers to 2-int vectors in AS(1)/AS(3)
typedef int v2i_ __attribute__((vector_size(8)));
typedef __attribute__((address_space(1))) v2i_* gptr_b64;
typedef __attribute__((address_space(3))) v2i_* lptr_b64;
typedef __attribute__((address_space(3))) void  lvoid_t;

__device__ __forceinline__ float sigmoidf_(float x) {
    return 1.0f / (1.0f + __expf(-x));
}

// One workgroup (8 wave32) per batch element.
__global__ __launch_bounds__(256)
void yolo_block_kernel(const float* __restrict__ preds,
                       const float* __restrict__ gt_boxes,
                       const int*   __restrict__ gt_labels,
                       const unsigned char* __restrict__ gt_valid,
                       float* __restrict__ accum)
{
    __shared__ float lds_preds[PRED_ELEMS];           // 17640 B
    __shared__ float lds_obj  [CELLS * B_ANCH];       // pm
    __shared__ float lds_conf [CELLS * B_ANCH];       // conf targets (iou_best)
    __shared__ float lds_coord[CELLS * B_ANCH * 4];   // coord targets
    __shared__ float lds_cls  [CELLS * C_CLS];        // one-hot class targets
    __shared__ float lds_wr   [8 * 8];                // per-wave partials [wave][acc]

    const int tid = threadIdx.x;
    const int bt  = blockIdx.x;
    const float* pb = preds + (size_t)bt * PRED_ELEMS;

    // ---- stage preds[bt] -> LDS via CDNA5 async copies (ASYNCcnt path) ----
    for (int i = tid; i < PRED_ELEMS / 2; i += 256) {      // 2205 B64 transfers
        const void* g = (const void*)(pb + 2 * i);
        void*       l = (void*)(lds_preds + 2 * i);
#if __has_builtin(__builtin_amdgcn_global_load_async_to_lds_b64)
        __builtin_amdgcn_global_load_async_to_lds_b64((gptr_b64)g, (lptr_b64)l, 0, 0);
#else
        unsigned lds_off = (unsigned)(__UINTPTR_TYPE__)(lvoid_t*)l;
        asm volatile("global_load_async_to_lds_b64 %0, %1, off"
                     :: "v"(lds_off), "v"(g) : "memory");
#endif
    }

    // zero target arrays while async copies are in flight
    for (int i = tid; i < CELLS * B_ANCH; i += 256) { lds_obj[i] = 0.f; lds_conf[i] = 0.f; }
    for (int i = tid; i < CELLS * B_ANCH * 4; i += 256) lds_coord[i] = 0.f;
    for (int i = tid; i < CELLS * C_CLS; i += 256) lds_cls[i] = 0.f;
    if (tid < 64) lds_wr[tid] = 0.f;

#if __has_builtin(__builtin_amdgcn_s_wait_asynccnt)
    __builtin_amdgcn_s_wait_asynccnt(0);
#else
    asm volatile("s_wait_asynccnt 0" ::: "memory");
#endif
    __syncthreads();

    // ---- phase A: build targets (wave 0, one thread per GT box) ----
    if (tid < N_GT) {
        const size_t gbase = (size_t)bt * N_GT + tid;
        if (gt_valid[gbase] != 0) {
            float x1 = gt_boxes[gbase * 4 + 0];
            float y1 = gt_boxes[gbase * 4 + 1];
            float x2 = gt_boxes[gbase * 4 + 2];
            float y2 = gt_boxes[gbase * 4 + 3];
            float cx = 0.5f * (x1 + x2), cy = 0.5f * (y1 + y2);
            float w  = fmaxf(x2 - x1, 1e-6f), h = fmaxf(y2 - y1, 1e-6f);
            int gi = min(max((int)floorf(cx * S_GRID), 0), S_GRID - 1);
            int gj = min(max((int)floorf(cy * S_GRID), 0), S_GRID - 1);
            int cell = gj * S_GRID + gi;
            const float* cp = lds_preds + cell * NCH;
            float area_g = (x2 - x1) * (y2 - y1);
            float best_iou = -1.f; int best = 0;
            #pragma unroll
            for (int bi = 0; bi < B_ANCH; ++bi) {
                float t0 = cp[bi*5+0], t1 = cp[bi*5+1], t2 = cp[bi*5+2], t3 = cp[bi*5+3];
                float px = (sigmoidf_(t0) + (float)gi) * (1.0f / S_GRID);
                float py = (sigmoidf_(t1) + (float)gj) * (1.0f / S_GRID);
                float pw = t2 * t2, ph = t3 * t3;
                float iw = fmaxf(fminf(px + 0.5f*pw, x2) - fmaxf(px - 0.5f*pw, x1), 0.f);
                float ih = fmaxf(fminf(py + 0.5f*ph, y2) - fmaxf(py - 0.5f*ph, y1), 0.f);
                float inter = iw * ih;
                float iou = inter / (pw * ph + area_g - inter + 1e-6f);
                if (iou > best_iou) { best_iou = iou; best = bi; }
            }
            int slot = cell * B_ANCH + best;
            lds_obj [slot] = 1.f;
            lds_conf[slot] = best_iou;
            lds_coord[slot*4+0] = cx * S_GRID - (float)gi;
            lds_coord[slot*4+1] = cy * S_GRID - (float)gj;
            lds_coord[slot*4+2] = sqrtf(w);
            lds_coord[slot*4+3] = sqrtf(h);
            lds_cls[cell * C_CLS + gt_labels[gbase]] = 1.f;
        }
    }
    __syncthreads();

    // ---- phase B: per-thread loss partials ----
    // acc: 0=coord 1=obj 2=noobj 3=cls 4=n_pos 5=n_neg 6=n_cell
    float acc[7] = {0.f, 0.f, 0.f, 0.f, 0.f, 0.f, 0.f};

    if (tid < CELLS * B_ANCH) {                 // 98 anchor slots, waves 0-3
        int cell = tid / B_ANCH, bi = tid % B_ANCH;
        const float* cp = lds_preds + cell * NCH + bi * 5;
        float pm = lds_obj[tid], nm = 1.f - pm;
        float sx = sigmoidf_(cp[0]), sy = sigmoidf_(cp[1]), so = sigmoidf_(cp[4]);
        const float* ct = lds_coord + tid * 4;
        float dx = sx - ct[0], dy = sy - ct[1];
        float dw = cp[2] - ct[2], dh = cp[3] - ct[3];
        acc[0] = pm * (dx*dx + dy*dy + dw*dw + dh*dh);
        float dob = so - lds_conf[tid];
        acc[1] = pm * dob * dob;
        acc[2] = nm * dob * dob;
        acc[4] = pm;
        acc[5] = nm;
    }
    if (tid >= 128 && tid < 128 + CELLS) {      // 49 cell softmaxes, waves 4-5
        int cell = tid - 128;
        float cm = (lds_obj[cell*B_ANCH] + lds_obj[cell*B_ANCH+1] > 0.f) ? 1.f : 0.f;
        const float* pc = lds_preds + cell * NCH + B_ANCH * 5;
        float mx = pc[0];
        for (int c = 1; c < C_CLS; ++c) mx = fmaxf(mx, pc[c]);
        float den = 0.f;
        for (int c = 0; c < C_CLS; ++c) den += __expf(pc[c] - mx);
        float inv = 1.f / den;
        float s = 0.f;
        for (int c = 0; c < C_CLS; ++c) {
            float d = __expf(pc[c] - mx) * inv - lds_cls[cell * C_CLS + c];
            s += d * d;
        }
        acc[3] = cm * s;
        acc[6] = cm;
    }

    // ---- reduction: shuffle within wave, then one WMMA across waves ----
    #pragma unroll
    for (int j = 0; j < 7; ++j) {
        float v = acc[j];
        for (int off = 16; off > 0; off >>= 1) v += __shfl_down(v, off, 32);
        acc[j] = v;
    }
    const int lane = tid & 31, wave = tid >> 5;
    if (lane == 0) {
        #pragma unroll
        for (int j = 0; j < 7; ++j) lds_wr[wave * 8 + j] = acc[j];
        lds_wr[wave * 8 + 7] = 0.f;
    }
    __syncthreads();

    if (wave == 0) {
        // A (16x32 f16): row m = accumulator m, K = wave index (K<8 live).
        // Lane L<16 holds row L halves 0..7 = K 0..7 (VGPR0..3), halves 8..15 = K 16..23.
        // B = all ones  =>  D[m][n] = sum_k A[m][k]  (grand total per accumulator).
        v16h a, bm;
        #pragma unroll
        for (int i = 0; i < 16; ++i) { a[i] = (_Float16)0.0f; bm[i] = (_Float16)1.0f; }
        if (lane < 16) {
            #pragma unroll
            for (int k = 0; k < 8; ++k)
                a[k] = (_Float16)((lane < 7) ? lds_wr[k * 8 + lane] : 0.f);
        }
        v8f c = {};
        v8f d = __builtin_amdgcn_wmma_f32_16x16x32_f16(
            /*neg_a=*/false, a, /*neg_b=*/false, bm,
            /*c_mod=*/(short)0, c, /*reuse_a=*/false, /*reuse_b=*/false);
        if (lane == 0) {
            // D layout: vgpr r, lane 0 -> M=r, N=0  => d[r] = total of accumulator r
            #pragma unroll
            for (int r = 0; r < 7; ++r) atomicAdd(&accum[r], d[r]);
        }
    }
}

__global__ void yolo_zero_kernel(float* __restrict__ accum) {
    if (threadIdx.x < 7) accum[threadIdx.x] = 0.f;
}

__global__ void yolo_finalize_kernel(const float* __restrict__ accum,
                                     float* __restrict__ out) {
    if (threadIdx.x == 0 && blockIdx.x == 0) {
        float coord = accum[0], obj = accum[1], noobj = accum[2], cls = accum[3];
        float n_pos  = fmaxf(accum[4], 1.f);
        float n_neg  = fmaxf(accum[5], 1.f);
        float n_cell = fmaxf(accum[6], 1.f);
        out[0] = LC_W * coord / n_pos + obj / n_pos
               + LN_W * noobj / n_neg + cls / n_cell;
    }
}

extern "C" void kernel_launch(void* const* d_in, const int* in_sizes, int n_in,
                              void* d_out, int out_size, void* d_ws, size_t ws_size,
                              hipStream_t stream)
{
    const float* preds            = (const float*)d_in[0];
    const float* gt_boxes         = (const float*)d_in[1];
    const int*   gt_labels        = (const int*)d_in[2];
    const unsigned char* gt_valid = (const unsigned char*)d_in[3];
    float* accum = (float*)d_ws;       // 7 fp32 accumulators
    float* out   = (float*)d_out;

    const int BT = in_sizes[0] / PRED_ELEMS;   // 8192

    yolo_zero_kernel<<<1, 32, 0, stream>>>(accum);
    yolo_block_kernel<<<BT, 256, 0, stream>>>(preds, gt_boxes, gt_labels, gt_valid, accum);
    yolo_finalize_kernel<<<1, 32, 0, stream>>>(accum, out);
}
